// YatCausalAttention_56367150793412
// MI455X (gfx1250) — matrix operand order repaired and checked by reading, not compiled, not run on hardware
//
#include <hip/hip_runtime.h>
#include <cstdint>

#define T_SEQ 2048
#define NBH   32          // B * H
#define HD    64
#define CDIM  1024
#define YEPS  1e-6f

typedef __attribute__((ext_vector_type(16))) __bf16         v16bf;
typedef __attribute__((ext_vector_type(16))) unsigned short v16us;
typedef __attribute__((ext_vector_type(8)))  float          v8f;
typedef __attribute__((ext_vector_type(4)))  unsigned int   v4u;
typedef __attribute__((ext_vector_type(8)))  int            v8i;
typedef __attribute__((ext_vector_type(4)))  int            v4i;

__device__ __forceinline__ unsigned short f2bf(float f) {
  unsigned int x = __builtin_bit_cast(unsigned int, f);
  unsigned int r = x + 0x7FFFu + ((x >> 16) & 1u);   // round-to-nearest-even
  return (unsigned short)(r >> 16);
}
__device__ __forceinline__ float bf2f(unsigned short u) {
  unsigned int x = ((unsigned int)u) << 16;
  return __builtin_bit_cast(float, x);
}
__device__ __forceinline__ int imin(int a, int b) { return a < b ? a : b; }

// ---------------------------------------------------------------------------
// TDM: DMA a [128 rows x 32 cols] bf16 tile (row stride = Kdim elements) from
// global memory into LDS at lds_addr, padded to an 80-byte LDS row pitch
// (64B data + 16B pad -> stride 40 ushorts, bank-conflict free for 16 lanes).
// D# packing per CDNA5 ISA ch.8 (group0 128b, group1 256b; groups 2/3 unused
// for 2-D tensors). Tracked by TENSORcnt.
// This toolchain exposes the 6-arg builtin:
//   (uint32x4 g0, int32x8 g1, int32x4 g2, int32x4 g3, int32x8 extra, i32 cpol)
__device__ __forceinline__ void tdm_load_tile_bf16(unsigned int lds_addr,
                                                   const unsigned short* gptr,
                                                   unsigned int Kdim) {
  unsigned long long ga = (unsigned long long)(const void*)gptr;
  v4u g0;
  g0[0] = 1u;                                            // count=1, user descriptor
  g0[1] = lds_addr;                                      // bits 63:32  lds_addr
  g0[2] = (unsigned int)ga;                              // bits 95:64  global_addr lo
  g0[3] = (unsigned int)((ga >> 32) & 0x1FFFFFFull)      // bits 120:96 global_addr hi
        | (2u << 30);                                    // bits 127:126 type=2 (image)
  v8i g1;
  g1[0] = (int)((1u << 16)                               // data_size = 1 -> 2 bytes
              | (1u << 20)                               // pad_enable
              | (3u << 22)                               // pad_interval: 16 DWORDs (64B)
              | (3u << 25));                             // pad_amount:   4 DWORDs (16B)
  g1[1] = (int)((Kdim & 0xFFFFu) << 16);                 // tensor_dim0[15:0]
  g1[2] = (int)(((Kdim >> 16) & 0xFFFFu)                 // tensor_dim0[31:16]
              | (128u << 16));                           // tensor_dim1[15:0] = 128
  g1[3] = (int)(32u << 16);                              // tensor_dim1 hi=0, tile_dim0=32
  g1[4] = (int)128u;                                     // tile_dim1=128, tile_dim2=0
  g1[5] = (int)Kdim;                                     // tensor_dim0_stride[31:0]
  g1[6] = 0;                                             // stride0 hi / dim1_stride lo
  g1[7] = 0;
  v4i z4 = {0, 0, 0, 0};
  v8i z8 = {0, 0, 0, 0, 0, 0, 0, 0};
  __builtin_amdgcn_tensor_load_to_lds(g0, g1, z4, z4, z8, 0);
}

// ---------------------------------------------------------------------------
// f32 -> bf16 row-major convert
__global__ void cvt_rowmajor(const float* __restrict__ src,
                             unsigned short* __restrict__ dst, int n) {
  int i = blockIdx.x * 256 + threadIdx.x;
  if (i < n) dst[i] = f2bf(src[i]);
}

// src [R][C] f32  ->  dst [C][R] bf16  (so GEMM B operand is [N][K], K contiguous)
__global__ void cvt_transpose(const float* __restrict__ src,
                              unsigned short* __restrict__ dst, int R, int C) {
  int i = blockIdx.x * 256 + threadIdx.x;
  if (i >= R * C) return;
  int r = i / C, c = i - r * C;
  dst[(size_t)c * R + r] = f2bf(src[i]);
}

// per-row squared norms of q and k (both [NBH*T_SEQ][HD] bf16)
__global__ void rowsq(const unsigned short* __restrict__ qb,
                      const unsigned short* __restrict__ kb,
                      float* __restrict__ qsq, float* __restrict__ ksq) {
  int i = blockIdx.x * 256 + threadIdx.x;
  if (i >= NBH * T_SEQ) return;
  float sq = 0.f, sk = 0.f;
  const unsigned short* qp = qb + (size_t)i * HD;
  const unsigned short* kp = kb + (size_t)i * HD;
#pragma unroll
  for (int j = 0; j < HD; ++j) {
    float q = bf2f(qp[j]); sq += q * q;
    float k = bf2f(kp[j]); sk += k * k;
  }
  qsq[i] = sq;
  ksq[i] = sk;
}

// ---------------------------------------------------------------------------
// Tiled bf16 GEMM with f32 accumulate:  C[M,N] = A[M,K] * B^T[N,K] + bias[N]
// Staging: double-buffered TDM tensor_load_to_lds issued by wave 0, overlapped
// with WMMA compute on the other buffer.
// mode 0: write f32 Cout[M][N]
// mode 1: QKV scatter -> q [BH,T,D] bf16, k [BH,T,D] bf16, v transposed [BH,D,T] bf16
__global__ __launch_bounds__(256)
void gemm_bf16(const unsigned short* __restrict__ A,
               const unsigned short* __restrict__ B,
               const float* __restrict__ bias,
               float* __restrict__ Cout,
               unsigned short* __restrict__ qb,
               unsigned short* __restrict__ kb,
               unsigned short* __restrict__ vT,
               int M, int N, int K, int mode) {
  __shared__ unsigned short lA[2][128 * 40];   // 128 rows x 32 K @ pitch 40
  __shared__ unsigned short lB[2][128 * 40];

  const int tid  = threadIdx.x;
  const int wave = tid >> 5;
  const int lane = tid & 31;
  const int n15  = lane & 15;
  const int half = lane >> 4;
  const int mw   = wave & 3;   // 4 wave-rows of 32
  const int nw   = wave >> 2;  // 2 wave-cols of 64
  const int m0   = blockIdx.x * 128;
  const int n0   = blockIdx.y * 128;

  v8f acc[2][4];
#pragma unroll
  for (int i = 0; i < 2; ++i)
#pragma unroll
    for (int j = 0; j < 4; ++j)
#pragma unroll
      for (int e = 0; e < 8; ++e) acc[i][j][e] = 0.f;

  // prologue: DMA first K tile into buffer 0
  if (wave == 0) {
    tdm_load_tile_bf16((unsigned int)(unsigned long long)(void*)&lA[0][0],
                       A + (size_t)m0 * K, (unsigned int)K);
    tdm_load_tile_bf16((unsigned int)(unsigned long long)(void*)&lB[0][0],
                       B + (size_t)n0 * K, (unsigned int)K);
    __builtin_amdgcn_s_wait_tensorcnt(0);
  }
  __syncthreads();

  const int nsteps = K >> 5;
  for (int s = 0; s < nsteps; ++s) {
    const int cur = s & 1;
    // kick off DMA of the next tile into the other buffer (overlaps with WMMA)
    if (wave == 0 && s + 1 < nsteps) {
      int k0n = (s + 1) << 5;
      tdm_load_tile_bf16((unsigned int)(unsigned long long)(void*)&lA[cur ^ 1][0],
                         A + (size_t)m0 * K + k0n, (unsigned int)K);
      tdm_load_tile_bf16((unsigned int)(unsigned long long)(void*)&lB[cur ^ 1][0],
                         B + (size_t)n0 * K + k0n, (unsigned int)K);
    }

    // A fragments (16x32): lane holds row m=lane&15; K = 16*(lane>=16)+j
    v16bf af[2];
#pragma unroll
    for (int mi = 0; mi < 2; ++mi) {
      int arow = mw * 32 + mi * 16 + n15;
      v16us t;
#pragma unroll
      for (int j = 0; j < 16; ++j) {
        int kk = j + half * 16;
        t[j] = lA[cur][arow * 40 + kk];
      }
      af[mi] = __builtin_bit_cast(v16bf, t);
    }
    // B fragments (32x16): lane holds column n=lane&15; K = 16*(lane>=16)+j
#pragma unroll
    for (int ni = 0; ni < 4; ++ni) {
      int brow = nw * 64 + ni * 16 + n15;
      v16us t;
#pragma unroll
      for (int j = 0; j < 16; ++j) t[j] = lB[cur][brow * 40 + half * 16 + j];
      v16bf bfv = __builtin_bit_cast(v16bf, t);
      acc[0][ni] = __builtin_amdgcn_wmma_f32_16x16x32_bf16(
          false, af[0], false, bfv, (short)0, acc[0][ni], false, false);
      acc[1][ni] = __builtin_amdgcn_wmma_f32_16x16x32_bf16(
          false, af[1], false, bfv, (short)0, acc[1][ni], false, false);
    }

    // make next buffer visible to everyone before the next iteration
    if (wave == 0 && s + 1 < nsteps) __builtin_amdgcn_s_wait_tensorcnt(0);
    __syncthreads();
  }

  // epilogue: C/D layout m = r + 8*(lane>=16), n = lane&15
#pragma unroll
  for (int mi = 0; mi < 2; ++mi)
#pragma unroll
    for (int ni = 0; ni < 4; ++ni)
#pragma unroll
      for (int r = 0; r < 8; ++r) {
        int mg = m0 + mw * 32 + mi * 16 + half * 8 + r;
        int ng = n0 + nw * 64 + ni * 16 + n15;
        float val = acc[mi][ni][r] + bias[ng];
        if (mode == 0) {
          Cout[(size_t)mg * N + ng] = val;
        } else {
          int which = ng >> 10;          // 0=q 1=k 2=v
          int cc = ng & 1023;
          int hh = cc >> 6, dd = cc & 63;
          int bb = mg >> 11, tt = mg & 2047;
          int bh = bb * 16 + hh;
          unsigned short bv = f2bf(val);
          if (which == 0)      qb[((size_t)bh * T_SEQ + tt) * HD + dd] = bv;
          else if (which == 1) kb[((size_t)bh * T_SEQ + tt) * HD + dd] = bv;
          else                 vT[((size_t)bh * HD + dd) * T_SEQ + tt] = bv;
        }
      }
}

// ---------------------------------------------------------------------------
// Fused causal "yat" flash attention.
// grid = (T/128, B*H); 8 waves/block; each wave owns 16 query rows.
// score = dot^2 / (q^2 + k^2 - 2*dot + eps), causal mask, online softmax, O += P*V.
__global__ __launch_bounds__(256)
void yat_flash(const unsigned short* __restrict__ qb,
               const unsigned short* __restrict__ kb,
               const unsigned short* __restrict__ vT,
               const float* __restrict__ qsq,
               const float* __restrict__ ksq,
               unsigned short* __restrict__ attnb) {
  __shared__ unsigned short lP[8][16 * 32];  // per-wave P re-swizzle tile

  const int bh   = blockIdx.y;
  const int b    = bh >> 4;
  const int h    = bh & 15;
  const int tid  = threadIdx.x;
  const int wave = tid >> 5;
  const int lane = tid & 31;
  const int n15  = lane & 15;
  const int half = lane >> 4;
  const int q0   = blockIdx.x * 128 + wave * 16;

  unsigned short* myP = &lP[wave][0];
  const unsigned short* qbase = qb + (size_t)bh * T_SEQ * HD;
  const unsigned short* kbase = kb + (size_t)bh * T_SEQ * HD;
  const unsigned short* vbase = vT + (size_t)bh * HD * T_SEQ;
  const float* qsqb = qsq + (size_t)bh * T_SEQ;
  const float* ksqb = ksq + (size_t)bh * T_SEQ;

  // Q fragments for both D chunks (A-frag swizzle)
  v16bf qf[2];
  {
    const unsigned short* qr = qbase + (size_t)(q0 + n15) * HD;
#pragma unroll
    for (int c = 0; c < 2; ++c) {
      v16us t;
#pragma unroll
      for (int j = 0; j < 16; ++j) {
        int kk = c * 32 + j + half * 16;
        t[j] = qr[kk];
      }
      qf[c] = __builtin_bit_cast(v16bf, t);
    }
  }

  float qsq_r[8];
#pragma unroll
  for (int r = 0; r < 8; ++r) qsq_r[r] = qsqb[q0 + half * 8 + r];

  float rm[8], rl[8];
#pragma unroll
  for (int r = 0; r < 8; ++r) { rm[r] = -__builtin_inff(); rl[r] = 0.f; }
  v8f o[4];
#pragma unroll
  for (int dd = 0; dd < 4; ++dd)
#pragma unroll
    for (int e = 0; e < 8; ++e) o[dd][e] = 0.f;

  const int kend = q0 + 16;  // causal: k <= q0+15
  for (int kt = 0; kt < kend; kt += 32) {
    // ---- S = Q @ K^T for 32 key rows (two 16-col tiles)
    v8f s0, s1;
#pragma unroll
    for (int e = 0; e < 8; ++e) { s0[e] = 0.f; s1[e] = 0.f; }
    const int kr0 = kt + n15;
    const int kr1 = imin(kt + 16 + n15, T_SEQ - 1);
#pragma unroll
    for (int c = 0; c < 2; ++c) {
      v16bf bf0 = *(const v16bf*)(kbase + (size_t)kr0 * HD + c * 32 + half * 16);
      v16bf bf1 = *(const v16bf*)(kbase + (size_t)kr1 * HD + c * 32 + half * 16);
      s0 = __builtin_amdgcn_wmma_f32_16x16x32_bf16(false, qf[c], false, bf0,
                                                   (short)0, s0, false, false);
      s1 = __builtin_amdgcn_wmma_f32_16x16x32_bf16(false, qf[c], false, bf1,
                                                   (short)0, s1, false, false);
    }

    // ---- yat score + causal mask + online softmax
    const float ks0 = ksqb[kr0];
    const float ks1 = ksqb[kr1];
    const int gk0 = kt + n15;
    const int gk1 = kt + 16 + n15;
#pragma unroll
    for (int r = 0; r < 8; ++r) {
      int gq = q0 + half * 8 + r;
      float d0 = s0[r], d1 = s1[r];
      float sc0 = (d0 * d0) / (qsq_r[r] + ks0 - 2.f * d0 + YEPS);
      float sc1 = (d1 * d1) / (qsq_r[r] + ks1 - 2.f * d1 + YEPS);
      if (gk0 > gq) sc0 = -__builtin_inff();
      if (gk1 > gq) sc1 = -__builtin_inff();

      float mt = fmaxf(sc0, sc1);
#pragma unroll
      for (int mask = 1; mask <= 8; mask <<= 1)
        mt = fmaxf(mt, __shfl_xor(mt, mask, 32));
      float mnew = fmaxf(rm[r], mt);

      float e0 = __expf(sc0 - mnew);   // exp(-inf) = 0 handles masked cols
      float e1 = __expf(sc1 - mnew);
      float ss = e0 + e1;
#pragma unroll
      for (int mask = 1; mask <= 8; mask <<= 1)
        ss += __shfl_xor(ss, mask, 32);

      float scale = __expf(rm[r] - mnew);
      rl[r] = rl[r] * scale + ss;
      rm[r] = mnew;
#pragma unroll
      for (int dd = 0; dd < 4; ++dd) o[dd][r] *= scale;

      // write P (C-layout) to per-wave LDS tile as bf16
      myP[(half * 8 + r) * 32 + n15]      = f2bf(e0);
      myP[(half * 8 + r) * 32 + 16 + n15] = f2bf(e1);
    }

    // ---- read back P as A-fragment (16x32); same-wave DS ops are in-order
    v16bf pf;
    {
      v16us t;
#pragma unroll
      for (int j = 0; j < 16; ++j) {
        int kk = j + half * 16;
        t[j] = myP[n15 * 32 + kk];
      }
      pf = __builtin_bit_cast(v16bf, t);
    }

    // ---- O += P @ V  (V stored transposed [D][T] -> contiguous-K B-frags)
#pragma unroll
    for (int dd = 0; dd < 4; ++dd) {
      v16bf vfrag = *(const v16bf*)(vbase + (size_t)(dd * 16 + n15) * T_SEQ +
                                    kt + half * 16);
      o[dd] = __builtin_amdgcn_wmma_f32_16x16x32_bf16(false, pf, false, vfrag,
                                                      (short)0, o[dd], false, false);
    }
  }

  // ---- normalize and store attention output as bf16 [B*T][C]
#pragma unroll
  for (int dd = 0; dd < 4; ++dd)
#pragma unroll
    for (int r = 0; r < 8; ++r) {
      int tt = q0 + half * 8 + r;
      float val = o[dd][r] / rl[r];
      attnb[((size_t)b * T_SEQ + tt) * CDIM + h * HD + dd * 16 + n15] = f2bf(val);
    }
}

// ---------------------------------------------------------------------------
extern "C" void kernel_launch(void* const* d_in, const int* in_sizes, int n_in,
                              void* d_out, int out_size, void* d_ws, size_t ws_size,
                              hipStream_t stream) {
  const float* x     = (const float*)d_in[0];
  const float* w_qkv = (const float*)d_in[1];
  const float* b_qkv = (const float*)d_in[2];
  const float* w_out = (const float*)d_in[3];
  const float* b_out = (const float*)d_in[4];
  float* out = (float*)d_out;

  char* ws = (char*)d_ws;
  size_t off = 0;
  auto take = [&](size_t bytes) -> char* {
    char* p = ws + off;
    off += (bytes + 255) & ~(size_t)255;
    return p;
  };

  unsigned short* xb    = (unsigned short*)take((size_t)4096 * CDIM * 2);
  unsigned short* wqkvT = (unsigned short*)take((size_t)3 * CDIM * CDIM * 2);
  unsigned short* woT   = (unsigned short*)take((size_t)CDIM * CDIM * 2);
  unsigned short* qb    = (unsigned short*)take((size_t)NBH * T_SEQ * HD * 2);
  unsigned short* kbuf  = (unsigned short*)take((size_t)NBH * T_SEQ * HD * 2);
  unsigned short* vT    = (unsigned short*)take((size_t)NBH * T_SEQ * HD * 2);
  float* qsq = (float*)take((size_t)NBH * T_SEQ * 4);
  float* ksq = (float*)take((size_t)NBH * T_SEQ * 4);
  unsigned short* attnb = xb;  // alias: xb fully consumed by gemm1 before flash writes

  cvt_rowmajor<<<(4096 * CDIM) / 256, 256, 0, stream>>>(x, xb, 4096 * CDIM);
  cvt_transpose<<<(CDIM * 3 * CDIM) / 256, 256, 0, stream>>>(w_qkv, wqkvT, CDIM, 3 * CDIM);
  cvt_transpose<<<(CDIM * CDIM) / 256, 256, 0, stream>>>(w_out, woT, CDIM, CDIM);

  // qkv = x @ w_qkv + b_qkv, scattered to q/k/vT
  gemm_bf16<<<dim3(4096 / 128, 3072 / 128), 256, 0, stream>>>(
      xb, wqkvT, b_qkv, nullptr, qb, kbuf, vT, 4096, 3072, CDIM, 1);

  rowsq<<<(NBH * T_SEQ) / 256, 256, 0, stream>>>(qb, kbuf, qsq, ksq);

  yat_flash<<<dim3(T_SEQ / 128, NBH), 256, 0, stream>>>(qb, kbuf, vT, qsq, ksq, attnb);

  // out = attn @ w_out + b_out  (f32)
  gemm_bf16<<<dim3(4096 / 128, 1024 / 128), 256, 0, stream>>>(
      attnb, woT, b_out, out, nullptr, nullptr, nullptr, 4096, 1024, CDIM, 0);
}
